// Decoder_18622978196085
// MI455X (gfx1250) — matrix operand order repaired and checked by reading, not compiled, not run on hardware
//
#include <hip/hip_runtime.h>
#include <math.h>

// ---------------------------------------------------------------------------
// 2-layer LSTM greedy decoder for MI455X (gfx1250).
//
// Bandwidth analysis: weights = 16384*(512+3*4096)*4B = 839 MB fp32 > 192 MB
// L2, so each of the 15 steps re-streams HBM.  fp32: 12.6 GB -> ~540us at
// 23.3 TB/s.  bf16 weights (one-time convert into d_ws) halve per-step
// traffic: ~54us convert + 15*18us = ~330us.  Compute is negligible, so we
// run the GEMVs on the matrix pipe: V_WMMA_F32_16X16X32_BF16 with
//   A = 16 rows x 32 K of W          (ISA 7.12.2 16-bit A layout)
//   B = input-vector slice broadcast to all 16 columns
// so D(:,n) is identical for all n; lanes 0 and 16 hold the 16 row dots.
// ---------------------------------------------------------------------------

#define EDIM 512
#define HDIM 4096
#define GDIM (4 * HDIM)   // 16384 gate rows
#define TSTEPS 15
#define KSPLIT 2          // deterministic K-partitioned partial sums

typedef __attribute__((ext_vector_type(16))) __bf16 v16bf;
typedef __attribute__((ext_vector_type(8)))  float  v8f;

union ABPack { v16bf v; uint4 q[2]; unsigned int u[8]; };

__device__ __forceinline__ unsigned short f2bf(float f) {
  unsigned int u = __float_as_uint(f);
  u += 0x7FFFu + ((u >> 16) & 1u);          // round-to-nearest-even
  return (unsigned short)(u >> 16);
}
__device__ __forceinline__ unsigned int pack2(float lo, float hi) {
  return (unsigned int)f2bf(lo) | ((unsigned int)f2bf(hi) << 16);
}

// --- weight segment loader: 8 consecutive bf16 elements into one uint4 -----
template <typename WT> struct WLoad;
template <> struct WLoad<unsigned short> {           // bf16-in-workspace path
  static __device__ __forceinline__ void seg8(const unsigned short* p, uint4& o) {
    o = *(const uint4*)p;                            // global_load_b128
  }
};
template <> struct WLoad<float> {                    // fp32 fallback path
  static __device__ __forceinline__ void seg8(const float* p, uint4& o) {
    float4 f0 = *(const float4*)p;
    float4 f1 = *(const float4*)(p + 4);
    o.x = pack2(f0.x, f0.y); o.y = pack2(f0.z, f0.w);
    o.z = pack2(f1.x, f1.y); o.w = pack2(f1.z, f1.w);
  }
};

// ---------------------------------------------------------------------------
// GEMV over the concatenated [Wa | Wb] K-range: parts[split][G] partial sums.
// One wave owns a 16-row tile; 256 threads = 8 waves/block; grid.x = G/128,
// grid.y = KSPLIT  -> 2048 waves streaming HBM.
// A per-lane data (16-bit 16x32 layout): row M = lane&15,
//   K segs [k + hi*8 .. +7] and [k + 16 + hi*8 .. +7], hi = lane>>4.
// B (columns all = v slice): lane needs v[k + hi*8 .. +7] and v[k + 16 + hi*8 .. +7].
// D layout: lane n<16 VGPR r = D(r,n); lane n+16 VGPR r = D(8+r,n);
//   -> lanes with (lane&15)==0 write rows tile + hi*8 .. +7.
// ---------------------------------------------------------------------------
template <typename WT>
__global__ __launch_bounds__(256)
void gemv_wmma(const WT* __restrict__ Wa, int Ka,
               const WT* __restrict__ Wb, int Kb,
               const unsigned short* __restrict__ va,
               const unsigned short* __restrict__ vb,
               float* __restrict__ parts)
{
  const int lane    = threadIdx.x & 31;
  const int wave    = threadIdx.x >> 5;
  const int rowTile = (blockIdx.x * 8 + wave) * 16;
  const int hi      = lane >> 4;
  const long row    = rowTile + (lane & 15);

  const int K      = Ka + Kb;
  const int kchunk = K / KSPLIT;            // always a multiple of 32 here
  const int k0     = blockIdx.y * kchunk;
  const int k1     = k0 + kchunk;

  v8f acc = {0.f, 0.f, 0.f, 0.f, 0.f, 0.f, 0.f, 0.f};

  // ---- segment overlapping Wa ([0,Ka)) ----
  {
    const int ea = (k1 < Ka) ? k1 : Ka;
    const WT* wr = Wa + row * (long)Ka;
    for (int k = k0; k < ea; k += 32) {
      __builtin_prefetch(wr + k + 256, 0, 1);   // global_prefetch, stream ahead
      ABPack a, b;
      WLoad<WT>::seg8(wr + k + hi * 8,      a.q[0]);
      WLoad<WT>::seg8(wr + k + hi * 8 + 16, a.q[1]);
      const unsigned short* bp = va + k + hi * 8;
      b.q[0] = *(const uint4*)bp;
      b.q[1] = *(const uint4*)(bp + 16);
      acc = __builtin_amdgcn_wmma_f32_16x16x32_bf16(
                false, a.v, false, b.v, (short)0, acc, false, false);
    }
  }
  // ---- segment overlapping Wb ([Ka,K)) ----
  {
    const int sb = (k0 > Ka) ? k0 : Ka;
    const WT* wr = Wb + row * (long)Kb;
    for (int k = sb; k < k1; k += 32) {
      const int kk = k - Ka;
      __builtin_prefetch(wr + kk + 256, 0, 1);
      ABPack a, b;
      WLoad<WT>::seg8(wr + kk + hi * 8,      a.q[0]);
      WLoad<WT>::seg8(wr + kk + hi * 8 + 16, a.q[1]);
      const unsigned short* bp = vb + kk + hi * 8;
      b.q[0] = *(const uint4*)bp;
      b.q[1] = *(const uint4*)(bp + 16);
      acc = __builtin_amdgcn_wmma_f32_16x16x32_bf16(
                false, a.v, false, b.v, (short)0, acc, false, false);
    }
  }

  if ((lane & 15) == 0) {                   // column 0 of D holds the dots
    const size_t idx = (size_t)blockIdx.y * GDIM + rowTile + hi * 8;
    *(float4*)(parts + idx)     = make_float4(acc[0], acc[1], acc[2], acc[3]);
    *(float4*)(parts + idx + 4) = make_float4(acc[4], acc[5], acc[6], acc[7]);
  }
}

__device__ __forceinline__ float sigmoidf(float x) {
  return 1.f / (1.f + expf(-x));
}

// ---------------------------------------------------------------------------
// Layer-0 cell: sum K-split partials + biases, gate nonlinearity, update c,
// emit h as bf16 (feeds the next WMMA GEMV as the B vector).
// ---------------------------------------------------------------------------
__global__ __launch_bounds__(256)
void lstm_cell0(const float* __restrict__ parts,
                const float* __restrict__ b_ih,
                const float* __restrict__ b_hh,
                float* __restrict__ c,
                unsigned short* __restrict__ hb)
{
  const int j = blockIdx.x * blockDim.x + threadIdx.x;
  float gi = b_ih[j]            + b_hh[j];
  float gf = b_ih[HDIM + j]     + b_hh[HDIM + j];
  float gg = b_ih[2 * HDIM + j] + b_hh[2 * HDIM + j];
  float go = b_ih[3 * HDIM + j] + b_hh[3 * HDIM + j];
#pragma unroll
  for (int s = 0; s < KSPLIT; ++s) {
    const float* p = parts + (size_t)s * GDIM;
    gi += p[j]; gf += p[HDIM + j]; gg += p[2 * HDIM + j]; go += p[3 * HDIM + j];
  }
  const float i = sigmoidf(gi), f = sigmoidf(gf);
  const float g = tanhf(gg),    o = sigmoidf(go);
  const float cn = f * c[j] + i * g;
  c[j] = cn;
  hb[j] = f2bf(o * tanhf(cn));
}

// ---------------------------------------------------------------------------
// Layer-1 cell fused with argmax (first-max-index, matching jnp.argmax) and
// embedding gather for the next step's input.  One workgroup, fixed-order
// LDS tree reduction -> bit-deterministic across replays.
// ---------------------------------------------------------------------------
__global__ __launch_bounds__(1024)
void lstm_cell1_argmax(const float* __restrict__ parts,
                       const float* __restrict__ b_ih,
                       const float* __restrict__ b_hh,
                       float* __restrict__ c,
                       unsigned short* __restrict__ hb,
                       float* __restrict__ outRow,
                       const float* __restrict__ embed,
                       unsigned short* __restrict__ xb)
{
  __shared__ float sval[1024];
  __shared__ int   sidx[1024];
  const int tid = threadIdx.x;

  float best = -INFINITY;
  int   bidx = 0;
#pragma unroll
  for (int kk = 0; kk < HDIM / 1024; ++kk) {
    const int j = tid + kk * 1024;
    float gi = b_ih[j]            + b_hh[j];
    float gf = b_ih[HDIM + j]     + b_hh[HDIM + j];
    float gg = b_ih[2 * HDIM + j] + b_hh[2 * HDIM + j];
    float go = b_ih[3 * HDIM + j] + b_hh[3 * HDIM + j];
#pragma unroll
    for (int s = 0; s < KSPLIT; ++s) {
      const float* p = parts + (size_t)s * GDIM;
      gi += p[j]; gf += p[HDIM + j]; gg += p[2 * HDIM + j]; go += p[3 * HDIM + j];
    }
    const float i = sigmoidf(gi), f = sigmoidf(gf);
    const float g = tanhf(gg),    o = sigmoidf(go);
    const float cn = f * c[j] + i * g;
    c[j] = cn;
    const float hn = o * tanhf(cn);
    hb[j]     = f2bf(hn);
    outRow[j] = hn;                         // this step's output row
    if (hn > best) { best = hn; bidx = j; } // strict > keeps earliest j
  }
  sval[tid] = best; sidx[tid] = bidx;
  __syncthreads();
  for (int s = 512; s > 0; s >>= 1) {
    if (tid < s) {
      const float v2 = sval[tid + s];
      const int   i2 = sidx[tid + s];
      if (v2 > sval[tid] || (v2 == sval[tid] && i2 < sidx[tid])) {
        sval[tid] = v2; sidx[tid] = i2;
      }
    }
    __syncthreads();
  }
  const int tok = sidx[0];
  if (tid < EDIM) xb[tid] = f2bf(embed[(size_t)tok * EDIM + tid]);
}

__global__ __launch_bounds__(256)
void init_state(const float* __restrict__ feat, unsigned short* __restrict__ xb,
                unsigned short* __restrict__ h0b, unsigned short* __restrict__ h1b,
                float* __restrict__ c0, float* __restrict__ c1)
{
  const int j = blockIdx.x * blockDim.x + threadIdx.x;
  if (j < EDIM) xb[j] = f2bf(feat[j]);
  if (j < HDIM) { h0b[j] = 0; h1b[j] = 0; c0[j] = 0.f; c1[j] = 0.f; }
}

__global__ __launch_bounds__(256)
void convert_bf16(const float* __restrict__ src, unsigned short* __restrict__ dst,
                  long n)
{
  const long i = ((long)blockIdx.x * blockDim.x + threadIdx.x) * 4;
  if (i < n) {
    const float4 f = *(const float4*)(src + i);
    uint2 o; o.x = pack2(f.x, f.y); o.y = pack2(f.z, f.w);
    *(uint2*)(dst + i) = o;
  }
}

// ---------------------------------------------------------------------------
extern "C" void kernel_launch(void* const* d_in, const int* in_sizes, int n_in,
                              void* d_out, int out_size, void* d_ws, size_t ws_size,
                              hipStream_t stream)
{
  const float* feat  = (const float*)d_in[0];
  const float* embed = (const float*)d_in[1];
  const float* Wih0  = (const float*)d_in[2];
  const float* Whh0  = (const float*)d_in[3];
  const float* bih0  = (const float*)d_in[4];
  const float* bhh0  = (const float*)d_in[5];
  const float* Wih1  = (const float*)d_in[6];
  const float* Whh1  = (const float*)d_in[7];
  const float* bih1  = (const float*)d_in[8];
  const float* bhh1  = (const float*)d_in[9];
  float* out = (float*)d_out;

  // workspace carve-up: small state first, bf16 weight cache after
  char*  w   = (char*)d_ws;
  size_t off = 0;
  auto carve = [&](size_t bytes) -> void* {
    void* p = w + off;
    off = (off + bytes + 255) & ~(size_t)255;
    return p;
  };
  unsigned short* xb    = (unsigned short*)carve(EDIM * 2);
  unsigned short* h0b   = (unsigned short*)carve(HDIM * 2);
  unsigned short* h1b   = (unsigned short*)carve(HDIM * 2);
  float*          c0    = (float*)carve(HDIM * 4);
  float*          c1    = (float*)carve(HDIM * 4);
  float*          parts = (float*)carve((size_t)KSPLIT * GDIM * 4);
  unsigned short* wih0b = (unsigned short*)carve((size_t)GDIM * EDIM * 2);
  unsigned short* whh0b = (unsigned short*)carve((size_t)GDIM * HDIM * 2);
  unsigned short* wih1b = (unsigned short*)carve((size_t)GDIM * HDIM * 2);
  unsigned short* whh1b = (unsigned short*)carve((size_t)GDIM * HDIM * 2);
  const bool bf16w = (ws_size >= off);      // fall back to fp32 streaming if small

  if (bf16w) {
    const long n0 = (long)GDIM * EDIM, n1 = (long)GDIM * HDIM;
    convert_bf16<<<(unsigned)((n0 / 4 + 255) / 256), 256, 0, stream>>>(Wih0, wih0b, n0);
    convert_bf16<<<(unsigned)((n1 / 4 + 255) / 256), 256, 0, stream>>>(Whh0, whh0b, n1);
    convert_bf16<<<(unsigned)((n1 / 4 + 255) / 256), 256, 0, stream>>>(Wih1, wih1b, n1);
    convert_bf16<<<(unsigned)((n1 / 4 + 255) / 256), 256, 0, stream>>>(Whh1, whh1b, n1);
  }
  init_state<<<16, 256, 0, stream>>>(feat, xb, h0b, h1b, c0, c1);

  const dim3 ggrid(GDIM / 128, KSPLIT);     // 128 x 2 blocks, 2048 waves
  for (int t = 0; t < TSTEPS; ++t) {
    if (bf16w)
      gemv_wmma<unsigned short><<<ggrid, 256, 0, stream>>>(
          wih0b, EDIM, whh0b, HDIM, xb, h0b, parts);
    else
      gemv_wmma<float><<<ggrid, 256, 0, stream>>>(
          Wih0, EDIM, Whh0, HDIM, xb, h0b, parts);
    lstm_cell0<<<HDIM / 256, 256, 0, stream>>>(parts, bih0, bhh0, c0, h0b);

    if (bf16w)
      gemv_wmma<unsigned short><<<ggrid, 256, 0, stream>>>(
          wih1b, HDIM, whh1b, HDIM, h0b, h1b, parts);
    else
      gemv_wmma<float><<<ggrid, 256, 0, stream>>>(
          Wih1, HDIM, Whh1, HDIM, h0b, h1b, parts);
    lstm_cell1_argmax<<<1, 1024, 0, stream>>>(
        parts, bih1, bhh1, c1, h1b, out + (size_t)t * HDIM, embed, xb);
  }
  (void)in_sizes; (void)n_in; (void)out_size;
}
